// MultiHeadAttention_30485677867871
// MI455X (gfx1250) — matrix operand order repaired and checked by reading, not compiled, not run on hardware
//
#include <hip/hip_runtime.h>

typedef __attribute__((ext_vector_type(16))) _Float16 v16h;
typedef __attribute__((ext_vector_type(8)))  _Float16 v8h;
typedef __attribute__((ext_vector_type(4)))  _Float16 v4h;
typedef __attribute__((ext_vector_type(8)))  float    v8f;
typedef __attribute__((ext_vector_type(4)))  float    v4f;
typedef unsigned int v4u __attribute__((ext_vector_type(4)));
typedef int          v8i __attribute__((ext_vector_type(8)));
typedef int          v4i __attribute__((ext_vector_type(4)));

constexpr int N  = 2048;
constexpr int D  = 1024;
constexpr int DK = 64;   // d_k per head

// ---------------------------------------------------------------------------
// WMMA helpers (gfx1250 wave32 layouts per CDNA5 ISA 7.12.2)
// ---------------------------------------------------------------------------
__device__ inline v16h load_a_frag(const _Float16* src, int ld, int lane) {
  int m = lane & 15, hi = lane >> 4;
  const _Float16* p = src + (size_t)m * ld + hi * 8;
  v8h c0 = *(const v8h*)(p);
  v8h c1 = *(const v8h*)(p + 16);
  v16h a;
#pragma unroll
  for (int i = 0; i < 8; ++i) { a[i] = c0[i]; a[i + 8] = c1[i]; }
  return a;
}

// B-matrix 32x16 f16 fragment: element(k,n) = src[n*ld + k];
// lane n = lane&15, k = (lane>>4)*8 + e (e<8), 16 + (lane>>4)*8 + (e-8) (e>=8).
__device__ inline v16h load_b_frag(const _Float16* src, int ld, int lane) {
  int n = lane & 15, kb = (lane >> 4) * 8;
  const _Float16* p = src + (size_t)n * ld + kb;
  v8h c0 = *(const v8h*)(p);
  v8h c1 = *(const v8h*)(p + 16);
  v16h b;
#pragma unroll
  for (int i = 0; i < 8; ++i) { b[i] = c0[i]; b[i + 8] = c1[i]; }
  return b;
}

__device__ inline v8f wmma_f16(v16h a, v16h b, v8f c) {
  return __builtin_amdgcn_wmma_f32_16x16x32_f16(false, a, false, b, (short)0, c,
                                                false, false);
}

// ---------------------------------------------------------------------------
// Tensor Data Mover: 2-D f16 tile (tile_w x tile_h) from global to LDS.
// Descriptor packing per CDNA5 ISA ch.8 (D# groups 0/1; groups 2/3 zero = 2D).
// This toolchain's builtin takes 6 args (…, int32x4 g2, int32x4 g3, int32x8, cpol).
// ---------------------------------------------------------------------------
__device__ inline void tdm_load_tile_f16(unsigned int lds_byte_addr,
                                         const void* gaddr,
                                         unsigned int tile_w,   // elements
                                         unsigned int tile_h,   // rows
                                         unsigned int row_stride) {  // elements
  unsigned long long ga = (unsigned long long)(uintptr_t)gaddr;
  v4u g0;
  g0[0] = 1u;                                   // count=1, user descriptor
  g0[1] = lds_byte_addr;                        // D#.lds_addr
  g0[2] = (unsigned int)(ga & 0xFFFFFFFFu);     // global_addr[31:0]
  g0[3] = (unsigned int)((ga >> 32) & 0x01FFFFFFu) | (2u << 30);  // [56:32]|type=2
  v8i g1;
  g1[0] = (int)(1u << 16);                      // data_size = 1 (2-byte elems)
  g1[1] = (int)((tile_w & 0xFFFFu) << 16);      // tensor_dim0[15:0] @ bits 63:48
  g1[2] = (int)(((tile_w >> 16) & 0xFFFFu) | ((tile_h & 0xFFFFu) << 16));
  g1[3] = (int)(((tile_h >> 16) & 0xFFFFu) | ((tile_w & 0xFFFFu) << 16));  // tile_dim0
  g1[4] = (int)(tile_h & 0xFFFFu);              // tile_dim1 (tile_dim2 = 0)
  g1[5] = (int)row_stride;                      // tensor_dim0_stride[31:0]
  g1[6] = 0;                                    // stride0[47:32], stride1[15:0]
  g1[7] = 0;                                    // stride1[47:16]
  v4i z4 = {0, 0, 0, 0};
  v8i z8 = {0, 0, 0, 0, 0, 0, 0, 0};
  __builtin_amdgcn_tensor_load_to_lds(g0, g1, z4, z4, z8, 0);
}

// ---------------------------------------------------------------------------
// f32 -> f16 conversion (vectorized x4)
// ---------------------------------------------------------------------------
__global__ void cvt_f32_to_f16(const float* __restrict__ src,
                               _Float16* __restrict__ dst, int n4) {
  int i = blockIdx.x * blockDim.x + threadIdx.x;
  if (i < n4) {
    v4f s = *(const v4f*)(src + (size_t)i * 4);
    v4h d;
#pragma unroll
    for (int j = 0; j < 4; ++j) d[j] = (_Float16)s[j];
    *(v4h*)(dst + (size_t)i * 4) = d;
  }
}

// ---------------------------------------------------------------------------
// Ping-pong pipelined GEMM core: acc[0..3] += X[16 x 1024] @ W^T[1024 x 64].
// Two named fragment sets, no copies -> distinct registers, loads overlap WMMA.
// ---------------------------------------------------------------------------
__device__ inline void gemm_core(const _Float16* __restrict__ Xrow,   // +row0*D
                                 const _Float16* __restrict__ Wcol,   // +col0*D
                                 int lane, v8f acc[4]) {
  v16h a0 = load_a_frag(Xrow + 0, D, lane);
  v16h b0[4];
#pragma unroll
  for (int t = 0; t < 4; ++t)
    b0[t] = load_b_frag(Wcol + (size_t)(t * 16) * D + 0, D, lane);
  v16h a1 = load_a_frag(Xrow + 32, D, lane);
  v16h b1[4];
#pragma unroll
  for (int t = 0; t < 4; ++t)
    b1[t] = load_b_frag(Wcol + (size_t)(t * 16) * D + 32, D, lane);

  for (int k0 = 0; k0 < D - 64; k0 += 64) {
    // consume set 0, immediately refill it for k0+64
#pragma unroll
    for (int t = 0; t < 4; ++t) acc[t] = wmma_f16(a0, b0[t], acc[t]);
    a0 = load_a_frag(Xrow + k0 + 64, D, lane);
#pragma unroll
    for (int t = 0; t < 4; ++t)
      b0[t] = load_b_frag(Wcol + (size_t)(t * 16) * D + k0 + 64, D, lane);

    // consume set 1 (its loads have been in flight one full stage), refill
#pragma unroll
    for (int t = 0; t < 4; ++t) acc[t] = wmma_f16(a1, b1[t], acc[t]);
    a1 = load_a_frag(Xrow + k0 + 96, D, lane);
#pragma unroll
    for (int t = 0; t < 4; ++t)
      b1[t] = load_b_frag(Wcol + (size_t)(t * 16) * D + k0 + 96, D, lane);
  }
#pragma unroll
  for (int t = 0; t < 4; ++t) acc[t] = wmma_f16(a0, b0[t], acc[t]);
#pragma unroll
  for (int t = 0; t < 4; ++t) acc[t] = wmma_f16(a1, b1[t], acc[t]);
}

// ---------------------------------------------------------------------------
// C = X[N x D] @ W^T + bias. f16 in, f32 acc, f16 out.
// transposed==1 stores C^T (Ct[col*N + row]) for the V projection.
// ---------------------------------------------------------------------------
__global__ __launch_bounds__(128) void proj_gemm(
    const _Float16* __restrict__ X, const _Float16* __restrict__ W,
    const float* __restrict__ bias, _Float16* __restrict__ Cout,
    int transposed) {
  int lane = threadIdx.x & 31;
  int wave = threadIdx.x >> 5;
  int row0 = blockIdx.y * 64 + wave * 16;
  int col0 = blockIdx.x * 64;

  v8f acc[4] = {};
  gemm_core(X + (size_t)row0 * D, W + (size_t)col0 * D, lane, acc);

  int nlane = lane & 15, hi = lane >> 4;
  if (!transposed) {
#pragma unroll
    for (int t = 0; t < 4; ++t) {
      int col = col0 + t * 16 + nlane;
      float bv = bias[col];
#pragma unroll
      for (int r = 0; r < 8; ++r) {
        int row = row0 + hi * 8 + r;
        Cout[(size_t)row * D + col] = (_Float16)(acc[t][r] + bv);
      }
    }
  } else {
#pragma unroll
    for (int t = 0; t < 4; ++t) {
      int col = col0 + t * 16 + nlane;
      float bv = bias[col];
      v8h o;
#pragma unroll
      for (int r = 0; r < 8; ++r) o[r] = (_Float16)(acc[t][r] + bv);
      *(v8h*)(Cout + (size_t)col * N + row0 + hi * 8) = o;  // 16B store
    }
  }
}

// ---------------------------------------------------------------------------
// Out = X[N x D] @ Wo^T + bo, f32 output (final projection).
// ---------------------------------------------------------------------------
__global__ __launch_bounds__(128) void out_gemm(
    const _Float16* __restrict__ X, const _Float16* __restrict__ W,
    const float* __restrict__ bias, float* __restrict__ Out) {
  int lane = threadIdx.x & 31;
  int wave = threadIdx.x >> 5;
  int row0 = blockIdx.y * 64 + wave * 16;
  int col0 = blockIdx.x * 64;

  v8f acc[4] = {};
  gemm_core(X + (size_t)row0 * D, W + (size_t)col0 * D, lane, acc);

  int nlane = lane & 15, hi = lane >> 4;
#pragma unroll
  for (int t = 0; t < 4; ++t) {
    int col = col0 + t * 16 + nlane;
    float bv = bias[col];
#pragma unroll
    for (int r = 0; r < 8; ++r) {
      int row = row0 + hi * 8 + r;
      Out[(size_t)row * D + col] = acc[t][r] + bv;
    }
  }
}

// ---------------------------------------------------------------------------
// Streaming causal attention, one wave per (16-query tile, head).
// K tiles (32 keys x 64 dims, f16) are staged into double-buffered LDS by the
// Tensor Data Mover; TENSORcnt waits synchronize; score B-frags read from LDS.
// V fragments stay register-pipelined from global (transposed Vt layout).
// ---------------------------------------------------------------------------
__global__ __launch_bounds__(32) void attn_kernel(
    const _Float16* __restrict__ Qp, const _Float16* __restrict__ Kp,
    const _Float16* __restrict__ Vt, _Float16* __restrict__ Oattn) {
  __shared__ __align__(32) _Float16 Ktile[2][32 * DK];  // 2 x 4KB
  __shared__ __align__(32) _Float16 Pbuf[16 * 32];      // 1KB

  int lane  = threadIdx.x & 31;
  int qBase = blockIdx.x * 16;
  int h     = blockIdx.y;
  int nlane = lane & 15, hi = lane >> 4;

  v16h qa0 = load_a_frag(Qp + (size_t)qBase * D + h * DK + 0,  D, lane);
  v16h qa1 = load_a_frag(Qp + (size_t)qBase * D + h * DK + 32, D, lane);

  v8f acc[4] = {};
  float Mrow[8], Srow[8];
#pragma unroll
  for (int r = 0; r < 8; ++r) { Mrow[r] = -__builtin_inff(); Srow[r] = 0.f; }

  const float sc = 0.125f;  // 1/sqrt(64)
  int nSteps = (qBase + 16 + 31) / 32;  // causal bound: keys <= qBase+15

  // TDM: kick off tile 0 into LDS buffer 0
  tdm_load_tile_f16((unsigned int)(uintptr_t)&Ktile[0][0],
                    Kp + (size_t)0 * D + h * DK, DK, 32, D);

  for (int jj = 0; jj < nSteps; ++jj) {
    int kk = jj * 32;
    // issue next K tile, then wait until the current one has landed
    if (jj + 1 < nSteps) {
      tdm_load_tile_f16((unsigned int)(uintptr_t)&Ktile[(jj + 1) & 1][0],
                        Kp + (size_t)(kk + 32) * D + h * DK, DK, 32, D);
      __builtin_amdgcn_s_wait_tensorcnt(1);  // all but the newest done
    } else {
      __builtin_amdgcn_s_wait_tensorcnt(0);
    }
    const _Float16* Kt = &Ktile[jj & 1][0];  // row-major [key][d], ld = 64

    // ---- scores: S[16q x 32keys] = Q @ K^T (from LDS) ----
    v8f s0 = {}, s1 = {};
    s0 = wmma_f16(qa0, load_b_frag(Kt + 0,           DK, lane), s0);
    s0 = wmma_f16(qa1, load_b_frag(Kt + 32,          DK, lane), s0);
    s1 = wmma_f16(qa0, load_b_frag(Kt + 16 * DK + 0, DK, lane), s1);
    s1 = wmma_f16(qa1, load_b_frag(Kt + 16 * DK + 32, DK, lane), s1);

    // V fragments for this tile (fly during the softmax below)
    v16h vb[4];
#pragma unroll
    for (int t = 0; t < 4; ++t)
      vb[t] = load_b_frag(Vt + (size_t)(h * DK + t * 16) * N + kk, N, lane);

    // ---- scale + causal mask ----
    bool needMask = (kk + 31 > qBase);
    float tmax[8];
#pragma unroll
    for (int r = 0; r < 8; ++r) {
      int row = qBase + hi * 8 + r;
      float a0 = s0[r] * sc, a1 = s1[r] * sc;
      if (needMask) {
        if (kk + nlane > row)      a0 = -__builtin_inff();
        if (kk + 16 + nlane > row) a1 = -__builtin_inff();
      }
      s0[r] = a0; s1[r] = a1;
      tmax[r] = fmaxf(a0, a1);
    }
#pragma unroll
    for (int r = 0; r < 8; ++r)
      for (int off = 1; off < 16; off <<= 1)
        tmax[r] = fmaxf(tmax[r], __shfl_xor(tmax[r], off, 16));

    // ---- online softmax update ----
    float scaleArr[8], psum[8];
#pragma unroll
    for (int r = 0; r < 8; ++r) {
      float newM = fmaxf(Mrow[r], tmax[r]);
      scaleArr[r] = __expf(Mrow[r] - newM);
      float p0 = __expf(s0[r] - newM);
      float p1 = __expf(s1[r] - newM);
      s0[r] = p0; s1[r] = p1;
      psum[r] = p0 + p1;
      Mrow[r] = newM;
    }
#pragma unroll
    for (int r = 0; r < 8; ++r)
      for (int off = 1; off < 16; off <<= 1)
        psum[r] += __shfl_xor(psum[r], off, 16);
#pragma unroll
    for (int r = 0; r < 8; ++r) Srow[r] = Srow[r] * scaleArr[r] + psum[r];
#pragma unroll
    for (int t = 0; t < 4; ++t)
#pragma unroll
      for (int r = 0; r < 8; ++r) acc[t][r] *= scaleArr[r];

    // ---- re-stripe P (C-layout -> A-layout) through LDS ----
    __syncthreads();
#pragma unroll
    for (int r = 0; r < 8; ++r) {
      int row = hi * 8 + r;
      Pbuf[row * 32 + nlane]      = (_Float16)s0[r];
      Pbuf[row * 32 + 16 + nlane] = (_Float16)s1[r];
    }
    __syncthreads();
    v16h pa = load_a_frag(Pbuf, 32, lane);

    // ---- O += P @ V ----
#pragma unroll
    for (int t = 0; t < 4; ++t) acc[t] = wmma_f16(pa, vb[t], acc[t]);
    __syncthreads();
  }

  // ---- normalize + store ----
  float inv[8];
#pragma unroll
  for (int r = 0; r < 8; ++r) inv[r] = 1.0f / Srow[r];
#pragma unroll
  for (int t = 0; t < 4; ++t) {
    int dcol = h * DK + t * 16 + nlane;
#pragma unroll
    for (int r = 0; r < 8; ++r) {
      int row = qBase + hi * 8 + r;
      Oattn[(size_t)row * D + dcol] = (_Float16)(acc[t][r] * inv[r]);
    }
  }
}

// ---------------------------------------------------------------------------
// Host-side orchestration
// ---------------------------------------------------------------------------
extern "C" void kernel_launch(void* const* d_in, const int* in_sizes, int n_in,
                              void* d_out, int out_size, void* d_ws,
                              size_t ws_size, hipStream_t stream) {
  const float* q  = (const float*)d_in[0];
  const float* k  = (const float*)d_in[1];
  const float* v  = (const float*)d_in[2];
  const float* Wq = (const float*)d_in[3];
  const float* bq = (const float*)d_in[4];
  const float* Wk = (const float*)d_in[5];
  const float* bk = (const float*)d_in[6];
  const float* Wv = (const float*)d_in[7];
  const float* bv = (const float*)d_in[8];
  const float* Wo = (const float*)d_in[9];
  const float* bo = (const float*)d_in[10];
  float* out = (float*)d_out;

  const size_t ND = (size_t)N * D;  // 2,097,152
  const size_t DD = (size_t)D * D;  // 1,048,576
  _Float16* w    = (_Float16*)d_ws;
  _Float16* q16  = w; w += ND;
  _Float16* k16  = w; w += ND;
  _Float16* v16  = w; w += ND;
  _Float16* Wq16 = w; w += DD;
  _Float16* Wk16 = w; w += DD;
  _Float16* Wv16 = w; w += DD;
  _Float16* Wo16 = w; w += DD;
  _Float16* Qp   = w; w += ND;
  _Float16* Kp   = w; w += ND;
  _Float16* Vt   = w; w += ND;   // transposed V projection [D, N]
  _Float16* Oat  = w; w += ND;

  {
    int nd4 = (int)(ND / 4), dd4 = (int)(DD / 4);
    int bND = (nd4 + 255) / 256, bDD = (dd4 + 255) / 256;
    cvt_f32_to_f16<<<bND, 256, 0, stream>>>(q,  q16,  nd4);
    cvt_f32_to_f16<<<bND, 256, 0, stream>>>(k,  k16,  nd4);
    cvt_f32_to_f16<<<bND, 256, 0, stream>>>(v,  v16,  nd4);
    cvt_f32_to_f16<<<bDD, 256, 0, stream>>>(Wq, Wq16, dd4);
    cvt_f32_to_f16<<<bDD, 256, 0, stream>>>(Wk, Wk16, dd4);
    cvt_f32_to_f16<<<bDD, 256, 0, stream>>>(Wv, Wv16, dd4);
    cvt_f32_to_f16<<<bDD, 256, 0, stream>>>(Wo, Wo16, dd4);
  }

  dim3 ggrid(D / 64, N / 64);  // (16, 32)
  proj_gemm<<<ggrid, 128, 0, stream>>>(q16, Wq16, bq, Qp, 0);
  proj_gemm<<<ggrid, 128, 0, stream>>>(k16, Wk16, bk, Kp, 0);
  proj_gemm<<<ggrid, 128, 0, stream>>>(v16, Wv16, bv, Vt, 1);

  attn_kernel<<<dim3(N / 16, D / DK), 32, 0, stream>>>(Qp, Kp, Vt, Oat);

  out_gemm<<<ggrid, 128, 0, stream>>>(Oat, Wo16, bo, out);
}
